// Attn_40656160424176
// MI455X (gfx1250) — matrix operand order repaired and checked by reading, not compiled, not run on hardware
//
#include <hip/hip_runtime.h>
#include <math.h>

#define B   8
#define CQ  256
#define CV  21
#define HW  56
#define N   3136      // HW*HW
#define KCHUNK 128
#define MW  7          // waves per block
#define MTILE (16*MW)  // 112 (m per block)
#define NTILE 64       // n per block
#define MT  (N/MTILE)  // 28
#define NTB (N/NTILE)  // 49
#define VCH 112        // n-chunk in finish kernel (28 chunks exact)

typedef _Float16 v8h  __attribute__((ext_vector_type(8)));
typedef _Float16 v16h __attribute__((ext_vector_type(16)));
typedef float    v8f  __attribute__((ext_vector_type(8)));

// -------- LDS byte-offset + async global->LDS copy helpers (CDNA5) --------
__device__ __forceinline__ unsigned lds_off(const void* p) {
    return (unsigned)(uintptr_t)(__attribute__((address_space(3))) const void*)p;
}
__device__ __forceinline__ void async_copy_b128(void* lds_dst, const void* gsrc) {
    unsigned l = lds_off(lds_dst);
    asm volatile("global_load_async_to_lds_b128 %0, %1, off"
                 :: "v"(l), "v"(gsrc) : "memory");
}
__device__ __forceinline__ void wait_async0() {
#if __has_builtin(__builtin_amdgcn_s_wait_asynccnt)
    __builtin_amdgcn_s_wait_asynccnt(0);
#else
    asm volatile("s_wait_asynccnt 0x0" ::: "memory");
#endif
}

// ---------------------------------------------------------------- utilities
__global__ __launch_bounds__(256) void zero_f32(float* __restrict__ p, int n) {
    int i = blockIdx.x * 256 + threadIdx.x;
    if (i < n) p[i] = 0.f;
}

// rq[b*CQ+c] = 1 / max(||q[b,c,:]||_2, 1e-12)   (norm over spatial axis)
__global__ __launch_bounds__(256) void qnorm_kernel(const float* __restrict__ q,
                                                    float* __restrict__ rq) {
    __shared__ float red[256];
    const int bc = blockIdx.x;
    const float* base = q + (size_t)bc * N;
    float s = 0.f;
    for (int i = threadIdx.x; i < N; i += 256) { float x = base[i]; s += x * x; }
    red[threadIdx.x] = s;
    __syncthreads();
    for (int off = 128; off > 0; off >>= 1) {
        if (threadIdx.x < off) red[threadIdx.x] += red[threadIdx.x + off];
        __syncthreads();
    }
    if (threadIdx.x == 0) rq[bc] = 1.f / fmaxf(sqrtf(red[0]), 1e-12f);
}

// rk[b*N+n] = 1 / max(||k[b,:,n]||_2, 1e-12)    (norm over channel axis)
__global__ __launch_bounds__(256) void knorm_kernel(const float* __restrict__ k,
                                                    float* __restrict__ rk) {
    const int idx = blockIdx.x * 256 + threadIdx.x;   // B*N exact
    const int b = idx / N, n = idx % N;
    const float* base = k + (size_t)b * CQ * N + n;
    float s = 0.f;
    #pragma unroll 8
    for (int c = 0; c < CQ; ++c) { float x = base[(size_t)c * N]; s += x * x; }
    rk[idx] = 1.f / fmaxf(sqrtf(s), 1e-12f);
}

// ------------------------------------------------- main WMMA energy/exp pass
// Block (7 waves) computes a 64(n) x 112(m) tile. Each wave: one 16-wide m
// strip, 4 accumulators along n (B fragment reused across 4 WMMAs).
__global__ __launch_bounds__(224) void attn_gemm_kernel(
    const float* __restrict__ q,   const float* __restrict__ kin,
    const float* __restrict__ gau, const float* __restrict__ rq,
    const float* __restrict__ rk,  float* __restrict__ attn,
    float* __restrict__ colsum)
{
    __shared__ __align__(16) _Float16 Ash[NTILE][KCHUNK + 8];
    __shared__ __align__(16) _Float16 Bsh[MTILE][KCHUNK + 8];

    const int tid  = threadIdx.x;
    const int wave = tid >> 5;
    const int lane = tid & 31;
    const int half = lane >> 4;
    const int l16  = lane & 15;

    int blk = blockIdx.x;                      // 0 .. B*NTB*MT-1
    const int b  = blk / (NTB * MT);
    int r  = blk - b * (NTB * MT);
    const int tn = r / MT;
    const int tm = r - tn * MT;
    const int n0 = tn * NTILE;
    const int m0 = tm * MTILE;

    const float* qb  = q   + (size_t)b * CQ * N;
    const float* kb  = kin + (size_t)b * CQ * N;
    const float* rqb = rq + b * CQ;

    v8f acc[4] = {};

    for (int k0 = 0; k0 < CQ; k0 += KCHUNK) {
        // Stage A tile (float4 reads): Ash[n][c] = fp16(q[b][k0+c][n0+n]*rq)
        for (int i = tid; i < (NTILE / 4) * KCHUNK; i += 224) {
            int n4 = (i & 15) * 4;
            int c  = i >> 4;
            float4 x = *(const float4*)&qb[(size_t)(k0 + c) * N + n0 + n4];
            float rs = rqb[k0 + c];
            Ash[n4 + 0][c] = (_Float16)(x.x * rs);
            Ash[n4 + 1][c] = (_Float16)(x.y * rs);
            Ash[n4 + 2][c] = (_Float16)(x.z * rs);
            Ash[n4 + 3][c] = (_Float16)(x.w * rs);
        }
        // Stage B tile (float4 reads): Bsh[m][c] = fp16(k[b][k0+c][m0+m])
        for (int i = tid; i < (MTILE / 4) * KCHUNK; i += 224) {   // 3584 = 224*16
            int m4 = (i % 28) * 4;
            int c  = i / 28;
            float4 x = *(const float4*)&kb[(size_t)(k0 + c) * N + m0 + m4];
            Bsh[m4 + 0][c] = (_Float16)x.x;
            Bsh[m4 + 1][c] = (_Float16)x.y;
            Bsh[m4 + 2][c] = (_Float16)x.z;
            Bsh[m4 + 3][c] = (_Float16)x.w;
        }
        __syncthreads();

        #pragma unroll
        for (int ks = 0; ks < KCHUNK; ks += 32) {
            // B 32x16 f16: lanes0-15 hold K[0..7,16..23] of col m=l16; lanes16-31 K[8..15,24..31]
            v8h blo = *(const v8h*)&Bsh[16 * wave + l16][ks + 8 * half];
            v8h bhi = *(const v8h*)&Bsh[16 * wave + l16][ks + 16 + 8 * half];
            v16h bf = __builtin_shufflevector(blo, bhi,
                0,1,2,3,4,5,6,7,8,9,10,11,12,13,14,15);
            #pragma unroll
            for (int t = 0; t < 4; ++t) {
                // A 16x32 f16: lanes0-15 K[0..7,16..23] of row M; lanes16-31 K[8..15,24..31]
                v8h alo = *(const v8h*)&Ash[16 * t + l16][ks + 8 * half];
                v8h ahi = *(const v8h*)&Ash[16 * t + l16][ks + 16 + 8 * half];
                v16h a = __builtin_shufflevector(alo, ahi,
                    0,1,2,3,4,5,6,7,8,9,10,11,12,13,14,15);
                acc[t] = __builtin_amdgcn_wmma_f32_16x16x32_f16(
                    false, a, false, bf, (short)0, acc[t], false, false);
            }
        }
        __syncthreads();
    }

    // Epilogue: C/D layout — VGPR j: lanes0-15 row +j, lanes16-31 row +j+8.
    const int m = m0 + 16 * wave + l16;
    const float rkm = rk[b * N + m];
    float* attnb = attn + (size_t)b * N * N;
    float part = 0.f;
    #pragma unroll
    for (int t = 0; t < 4; ++t) {
        #pragma unroll
        for (int j = 0; j < 8; ++j) {
            int n = n0 + 16 * t + j + 8 * half;
            float z  = acc[t][j] * rkm * gau[(size_t)n * N + m];
            float ex = __expf(z);
            attnb[(size_t)n * N + m] = ex;
            part += ex;
        }
    }
    part += __shfl_xor(part, 16);      // rows 0..7(+16t) with rows 8..15(+16t)
    if (half == 0) atomicAdd(&colsum[b * N + m], part);
}

// ---------------- fused: attn /= colsum (in place)  +  out = V @ attn ------
__global__ __launch_bounds__(256) void finish_kernel(
    const float* __restrict__ vin, const float* __restrict__ colsum,
    float* __restrict__ attn, float* __restrict__ out)
{
    __shared__ __align__(16) float vsh[CV][VCH];
    const int b = blockIdx.y;
    const int m = blockIdx.x * 256 + threadIdx.x;
    const float* vb = vin + (size_t)b * CV * N;
    float* ab = attn + (size_t)b * N * N;
    const float si = (m < N) ? (1.f / colsum[b * N + m]) : 0.f;

    float acc[CV];
    #pragma unroll
    for (int i = 0; i < CV; ++i) acc[i] = 0.f;

    for (int nb = 0; nb < N; nb += VCH) {                 // 28 chunks exact
        __syncthreads();
        // async-stage V[:, nb:nb+VCH] (raw fp32, 16B-aligned b128 transfers)
        for (int t = threadIdx.x; t < CV * (VCH / 4); t += 256) {
            int vv = t / (VCH / 4);
            int nn = t % (VCH / 4);
            async_copy_b128(&vsh[vv][4 * nn], &vb[(size_t)vv * N + nb + 4 * nn]);
        }
        wait_async0();
        __syncthreads();
        if (m < N) {
            if (nb + VCH < N)
                __builtin_prefetch(&ab[(size_t)(nb + VCH) * N + m], 0, 1);
            for (int nn = 0; nn < VCH; ++nn) {
                size_t o = (size_t)(nb + nn) * N + m;
                float a = ab[o] * si;     // normalized attention
                ab[o] = a;
                #pragma unroll
                for (int vv = 0; vv < CV; ++vv)
                    acc[vv] = fmaf(vsh[vv][nn], a, acc[vv]);
            }
        }
    }
    if (m < N) {
        #pragma unroll
        for (int vv = 0; vv < CV; ++vv)
            out[(size_t)b * CV * N + (size_t)vv * N + m] = acc[vv];
    }
}

// ---------------------------------------------------------------- launcher
extern "C" void kernel_launch(void* const* d_in, const int* in_sizes, int n_in,
                              void* d_out, int out_size, void* d_ws, size_t ws_size,
                              hipStream_t stream) {
    const float* q   = (const float*)d_in[0];  // [8,256,56,56]
    const float* k   = (const float*)d_in[1];  // [8,256,56,56]
    const float* v   = (const float*)d_in[2];  // [8,21,56,56]
    const float* gau = (const float*)d_in[3];  // [3136,3136]

    float* out  = (float*)d_out;               // 526848
    float* attn = out + (size_t)B * CV * N;    // 78675968

    float* ws = (float*)d_ws;
    float* rq = ws;                  // B*CQ  = 2048
    float* rk = ws + 2048;           // B*N   = 25088
    float* s  = ws + 2048 + 25088;   // B*N   = 25088  (column sums)

    zero_f32 <<<(B * N + 255) / 256, 256, 0, stream>>>(s, B * N);
    qnorm_kernel<<<B * CQ, 256, 0, stream>>>(q, rq);
    knorm_kernel<<<(B * N) / 256, 256, 0, stream>>>(k, rk);

    attn_gemm_kernel<<<B * NTB * MT, 224, 0, stream>>>(q, k, gau, rq, rk, attn, s);

    finish_kernel<<<dim3((N + 255) / 256, B), 256, 0, stream>>>(v, s, attn, out);
}
